// DirectVoxGO_23897198035512
// MI455X (gfx1250) — matrix operand order repaired and checked by reading, not compiled, not run on hardware
//
#include <hip/hip_runtime.h>
#include <hip/hip_fp16.h>

typedef __attribute__((ext_vector_type(16))) _Float16 v16h;
typedef __attribute__((ext_vector_type(8)))  float    v8f;

#define GRES      160
#define GSTRIDE   (GRES*GRES*GRES)          // 4,096,000
#define ACT_SHIFT (-4.595119850134589f)     // log(1/(1-0.01) - 1)
#define DELTA     0.5f
#define NFRAG_L0  16                        // 2 kc * 8 nt
#define NFRAG_L1  32                        // 4 kc * 8 nt
#define NFRAG_L2  4                         // 4 kc * 1 nt
#define NFRAG_TOT (NFRAG_L0 + NFRAG_L1 + NFRAG_L2)   // 52
#define MLP_WAVES 8
#define MLP_BLOCKS 1536

// ---------------- trilinear helpers ----------------
struct Corner {
  int base, dx, dy;
  float w000,w001,w010,w011,w100,w101,w110,w111;
};

__device__ __forceinline__ Corner corner_setup(float x, float y, float z) {
  float px = (x + 1.f) * 0.5f * (float)(GRES - 1);
  float py = (y + 1.f) * 0.5f * (float)(GRES - 1);
  float pz = (z + 1.f) * 0.5f * (float)(GRES - 1);
  px = fminf(fmaxf(px, 0.f), (float)(GRES - 1));
  py = fminf(fmaxf(py, 0.f), (float)(GRES - 1));
  pz = fminf(fmaxf(pz, 0.f), (float)(GRES - 1));
  int x0 = min(max((int)floorf(px), 0), GRES - 2);
  int y0 = min(max((int)floorf(py), 0), GRES - 2);
  int z0 = min(max((int)floorf(pz), 0), GRES - 2);
  float fx = px - (float)x0, fy = py - (float)y0, fz = pz - (float)z0;
  Corner c;
  c.base = (x0*GRES + y0)*GRES + z0;
  c.dx = GRES*GRES; c.dy = GRES;
  float gx = 1.f - fx, gy = 1.f - fy, gz = 1.f - fz;
  c.w000 = gx*gy*gz; c.w001 = gx*gy*fz;
  c.w010 = gx*fy*gz; c.w011 = gx*fy*fz;
  c.w100 = fx*gy*gz; c.w101 = fx*gy*fz;
  c.w110 = fx*fy*gz; c.w111 = fx*fy*fz;
  return c;
}

__device__ __forceinline__ float tri_gather(const float* __restrict__ g, const Corner& c) {
  return g[c.base            ]*c.w000 + g[c.base            +1]*c.w001
       + g[c.base       +c.dy]*c.w010 + g[c.base       +c.dy+1]*c.w011
       + g[c.base +c.dx      ]*c.w100 + g[c.base +c.dx      +1]*c.w101
       + g[c.base +c.dx +c.dy]*c.w110 + g[c.base +c.dx +c.dy+1]*c.w111;
}

// ---------------- ray segment bounds ----------------
__global__ void k_ray_bounds_init(int* __restrict__ starts, int* __restrict__ ends, int nrays) {
  const int r = blockIdx.x*blockDim.x + threadIdx.x;
  if (r < nrays) { starts[r] = -1; ends[r] = -1; }
}

__global__ void k_ray_bounds(const int* __restrict__ ray_id, int M,
                             int* __restrict__ starts, int* __restrict__ ends) {
  const int i = blockIdx.x*blockDim.x + threadIdx.x;
  if (i >= M) return;
  const int r = ray_id[i];
  if (i == 0   || ray_id[i-1] != r) starts[r] = i;
  if (i == M-1 || ray_id[i+1] != r) ends[r]   = i + 1;
}

// ---------------- per-ray transmittance: one wave32 per ray ----------------
// Lanes compute 32 points' alpha/log1ma in parallel; a 5-step shuffle prefix
// scan yields the exclusive cumulative log-transmittance; scalar carry chains
// 32-point chunks. 32x shorter dependent chain than thread-per-ray.
__global__ __launch_bounds__(256) void k_ray_scan(
    const float* __restrict__ dens, const float* __restrict__ xyz,
    const int* __restrict__ starts, const int* __restrict__ ends,
    float* __restrict__ wpt, float* __restrict__ out, int nrays) {
  const int gtid = blockIdx.x*blockDim.x + threadIdx.x;
  const int r    = gtid >> 5;
  const int lane = gtid & 31;
  if (r >= nrays) return;
  const int s = starts[r];
  float carry = 0.f;                         // cumulative log-transmittance
  if (s >= 0) {
    const int e = ends[r];
    for (int i0 = s; i0 < e; i0 += 32) {
      const int i = i0 + lane;
      float a = 0.f, l1 = 0.f;
      if (i < e) {
        const Corner c = corner_setup(xyz[3*i], xyz[3*i+1], xyz[3*i+2]);
        const float d  = tri_gather(dens, c) + ACT_SHIFT;
        const float sp = (d > 20.f) ? d : log1pf(__expf(d));   // softplus
        a  = 1.f - __expf(-sp * DELTA);
        l1 = log1pf(-fminf(a, 1.f - 1e-7f));
      }
      // inclusive prefix scan of l1 across the wave
      float sc = l1;
      #pragma unroll
      for (int d = 1; d < 32; d <<= 1) {
        const float o = __shfl_up(sc, d, 32);
        if (lane >= d) sc += o;
      }
      if (i < e) wpt[i] = __expf(carry + (sc - l1)) * a;       // exclusive + carry
      carry += __shfl(sc, 31, 32);                             // chunk total
    }
  }
  if (lane == 0) {
    const float av = __expf(carry);          // alphainv_last (1.0 for empty rays)
    out[3*r] = av; out[3*r+1] = av; out[3*r+2] = av;           // fresh base each call
  }
}

// ---------------- pack weights into WMMA B-fragment layout ----------------
// Fragment f, lane l, slot j (16 f16 per lane):  value = W[K][N] where
//   K = kc*32 + (l>>4)*8 + (j&7) + (j>>3)*16,  N = nt*16 + (l&15)   (ISA 7.12.2 B layout)
// Layer 0: b0 folded into padded row K==36 (feat[36] is set to 1.0).
__global__ void k_pack_weights(const float* __restrict__ W0, const float* __restrict__ b0,
                               const float* __restrict__ W1, const float* __restrict__ W2,
                               _Float16* __restrict__ wsW) {
  const int tid = blockIdx.x*blockDim.x + threadIdx.x;
  if (tid >= NFRAG_TOT*512) return;
  const int f    = tid >> 9;
  const int lane = (tid >> 4) & 31;
  const int j    = tid & 15;
  const int krow = ((lane >> 4) << 3) + (j & 7) + ((j >> 3) << 4);   // K offset inside 32-chunk
  const int n    = lane & 15;
  float v;
  if (f < NFRAG_L0) {                                   // W0: [36,128], K padded ->64
    const int kc = f >> 3, nt = f & 7;
    const int K = kc*32 + krow, N = nt*16 + n;
    v = (K < 36) ? W0[K*128 + N] : ((K == 36) ? b0[N] : 0.f);
  } else if (f < NFRAG_L0 + NFRAG_L1) {                 // W1: [128,128]
    const int ff = f - NFRAG_L0;
    const int kc = ff >> 3, nt = ff & 7;
    v = W1[(kc*32 + krow)*128 + nt*16 + n];
  } else {                                              // W2: [128,3] padded N->16
    const int kc = f - (NFRAG_L0 + NFRAG_L1);
    v = (n < 3) ? W2[(kc*32 + krow)*3 + n] : 0.f;
  }
  wsW[tid] = (_Float16)v;
}

// ---------------- fragment loaders ----------------
union Frag16 { v16h h; uint4 u[2]; };

__device__ __forceinline__ v16h load_bfrag(const _Float16* __restrict__ wsW, int f, int lane) {
  Frag16 b;
  const uint4* p = (const uint4*)(wsW + ((size_t)(f*32 + lane) << 4));  // 32B/lane, contiguous
  b.u[0] = p[0];
  b.u[1] = p[1];
  return b.h;
}

// ---------------- fused feat-build + 3-layer WMMA MLP + weighted scatter ----------------
__global__ __launch_bounds__(256, 1) void k_mlp(
    const float* __restrict__ k0g,   const float* __restrict__ xyz,
    const float* __restrict__ vdirs, const int*   __restrict__ ray_id,
    const float* __restrict__ b1,    const float* __restrict__ b2,
    const _Float16* __restrict__ wsW, const float* __restrict__ wpt,
    float* __restrict__ out, int M)
{
  __shared__ alignas(32) _Float16 hbuf[MLP_WAVES][16*128];  // feat[16x64] then h0/h1[16x128]
  __shared__ float kdiff[MLP_WAVES][16][3];
  const int wid  = threadIdx.x >> 5;
  const int lane = threadIdx.x & 31;
  const int t    = lane & 15;        // row / column index within 16-tile
  const int half = lane >> 4;        // lane-half selects K-subrange (ISA A/B layouts)
  const int kb   = half * 8;         // A-fragment K base within each 16-run
  _Float16* __restrict__ hb = hbuf[wid];

  const int T      = M >> 4;                      // 16-point tiles
  const int stride = gridDim.x * MLP_WAVES;
  const int iters  = (T + stride - 1) / stride;

  for (int it = 0; it < iters; ++it) {
    const int tile  = it*stride + blockIdx.x*MLP_WAVES + wid;
    const bool active = tile < T;
    __syncthreads();
    if (active) {
      // ---- Phase 1: build feat tile [16 x 64] f16 (+ diffuse [16 x 3] f32)
      int p = tile*16 + t;  if (p >= M) p = M - 1;
      const float x = xyz[3*p], y = xyz[3*p+1], z = xyz[3*p+2];
      const Corner c = corner_setup(x, y, z);
      _Float16* frow = hb + t*64;
      const int cbase = half * 6;                 // lane-halves split the 12 channels
      #pragma unroll
      for (int ch = 0; ch < 6; ++ch) {
        const int cc = cbase + ch;
        const float v = tri_gather(k0g + (size_t)cc*GSTRIDE, c);
        if (cc < 3) kdiff[wid][t][cc] = v;        // k0_diffuse
        else        frow[cc - 3] = (_Float16)v;   // feat[0..8] = k0_view
      }
      if (half == 0) {                            // feat[9..35] = vd, sin, cos
        const int rid = ray_id[p];
        const float vx = vdirs[3*rid], vy = vdirs[3*rid+1], vz = vdirs[3*rid+2];
        frow[9] = (_Float16)vx; frow[10] = (_Float16)vy; frow[11] = (_Float16)vz;
        const float vv[3] = {vx, vy, vz};
        #pragma unroll
        for (int d = 0; d < 3; ++d)
          #pragma unroll
          for (int fq = 0; fq < 4; ++fq) {
            const float e = vv[d] * (float)(1 << fq);
            frow[12 + d*4 + fq] = (_Float16)__sinf(e);
            frow[24 + d*4 + fq] = (_Float16)__cosf(e);
          }
      } else {
        frow[36] = (_Float16)1.f;                 // bias row selector (b0 in W0 K=36)
        #pragma unroll
        for (int q = 37; q < 64; ++q) frow[q] = (_Float16)0.f;
      }
    }
    __syncthreads();
    if (!active) continue;   // barrier sequence stays uniform across the block

    // ---- Layer 0: feat[16x64] @ W0' (bias folded), relu -> h0[16x128] in LDS
    Frag16 a0[2];
    #pragma unroll
    for (int kc = 0; kc < 2; ++kc) {              // A frag: two aligned ds_load_b128
      a0[kc].u[0] = *(const uint4*)(hb + t*64 + kc*32 + kb);
      a0[kc].u[1] = *(const uint4*)(hb + t*64 + kc*32 + 16 + kb);
    }
    #pragma unroll
    for (int nt = 0; nt < 8; ++nt) {
      v8f c = {0.f,0.f,0.f,0.f,0.f,0.f,0.f,0.f};  // bias folded into K=36 row
      c = __builtin_amdgcn_wmma_f32_16x16x32_f16(false, a0[0].h, false,
            load_bfrag(wsW, nt, lane),      (short)0, c, false, false);
      c = __builtin_amdgcn_wmma_f32_16x16x32_f16(false, a0[1].h, false,
            load_bfrag(wsW, 8 + nt, lane),  (short)0, c, false, false);
      #pragma unroll
      for (int r = 0; r < 8; ++r)                 // C layout: M = r + 8*half, N = nt*16+t
        hb[(r + half*8)*128 + nt*16 + t] = (_Float16)fmaxf(c[r], 0.f);
    }

    // ---- Layer 1: h0[16x128] @ W1 + b1, relu -> h1[16x128] (A frags hoisted before stores)
    Frag16 a1[4];
    #pragma unroll
    for (int kc = 0; kc < 4; ++kc) {
      a1[kc].u[0] = *(const uint4*)(hb + t*128 + kc*32 + kb);
      a1[kc].u[1] = *(const uint4*)(hb + t*128 + kc*32 + 16 + kb);
    }
    #pragma unroll
    for (int nt = 0; nt < 8; ++nt) {
      const float bias = b1[nt*16 + t];
      v8f c;
      #pragma unroll
      for (int r = 0; r < 8; ++r) c[r] = bias;
      #pragma unroll
      for (int kc = 0; kc < 4; ++kc)
        c = __builtin_amdgcn_wmma_f32_16x16x32_f16(false, a1[kc].h, false,
              load_bfrag(wsW, NFRAG_L0 + kc*8 + nt, lane), (short)0, c, false, false);
      #pragma unroll
      for (int r = 0; r < 8; ++r)
        hb[(r + half*8)*128 + nt*16 + t] = (_Float16)fmaxf(c[r], 0.f);
    }

    // ---- Layer 2: h1[16x128] @ W2 + b2 (N padded to 16)
    Frag16 a2[4];
    #pragma unroll
    for (int kc = 0; kc < 4; ++kc) {
      a2[kc].u[0] = *(const uint4*)(hb + t*128 + kc*32 + kb);
      a2[kc].u[1] = *(const uint4*)(hb + t*128 + kc*32 + 16 + kb);
    }
    v8f c2;
    const float bias2 = (t < 3) ? b2[t] : 0.f;
    #pragma unroll
    for (int r = 0; r < 8; ++r) c2[r] = bias2;
    #pragma unroll
    for (int kc = 0; kc < 4; ++kc)
      c2 = __builtin_amdgcn_wmma_f32_16x16x32_f16(false, a2[kc].h, false,
             load_bfrag(wsW, NFRAG_L0 + NFRAG_L1 + kc, lane), (short)0, c2, false, false);

    // ---- Output: sigmoid(logit + diffuse), weighted scatter to per-ray RGB
    if (t < 3) {
      #pragma unroll
      for (int r = 0; r < 8; ++r) {
        const int mm = r + half*8;
        const int p2 = tile*16 + mm;
        if (p2 < M) {
          const float logit = c2[r] + kdiff[wid][mm][t];
          const float rgb   = 1.f / (1.f + __expf(-logit));
          unsafeAtomicAdd(&out[(size_t)ray_id[p2]*3 + t], wpt[p2] * rgb);
        }
      }
    }
  }
}

// ---------------- host launch ----------------
extern "C" void kernel_launch(void* const* d_in, const int* in_sizes, int n_in,
                              void* d_out, int out_size, void* d_ws, size_t ws_size,
                              hipStream_t stream)
{
  const float* dens   = (const float*)d_in[0];
  const float* k0g    = (const float*)d_in[1];
  const float* xyz    = (const float*)d_in[2];
  const float* vdirs  = (const float*)d_in[3];
  const float* W0     = (const float*)d_in[4];
  const float* b0     = (const float*)d_in[5];
  const float* W1     = (const float*)d_in[6];
  const float* b1     = (const float*)d_in[7];
  const float* W2     = (const float*)d_in[8];
  const float* b2     = (const float*)d_in[9];
  const int*   ray_id = (const int*)d_in[10];
  const int M     = in_sizes[2] / 3;
  const int nrays = in_sizes[3] / 3;
  float* out = (float*)d_out;
  (void)n_in; (void)out_size; (void)ws_size;

  char* ws = (char*)d_ws;
  size_t o = 0;
  float*    wpt    = (float*)(ws + o); o += (size_t)M * sizeof(float);   o = (o + 255) & ~(size_t)255;
  int*      starts = (int*)(ws + o);   o += (size_t)nrays * sizeof(int); o = (o + 255) & ~(size_t)255;
  int*      ends   = (int*)(ws + o);   o += (size_t)nrays * sizeof(int); o = (o + 255) & ~(size_t)255;
  _Float16* wsW    = (_Float16*)(ws + o);

  k_ray_bounds_init<<<(nrays + 255)/256, 256, 0, stream>>>(starts, ends, nrays);
  k_ray_bounds     <<<(M + 255)/256,     256, 0, stream>>>(ray_id, M, starts, ends);
  k_pack_weights   <<<(NFRAG_TOT*512 + 255)/256, 256, 0, stream>>>(W0, b0, W1, W2, wsW);
  k_ray_scan       <<<((nrays*32) + 255)/256, 256, 0, stream>>>(dens, xyz, starts, ends, wpt, out, nrays);
  k_mlp            <<<MLP_BLOCKS, 256, 0, stream>>>(k0g, xyz, vdirs, ray_id,
                                                    b1, b2, wsW, wpt, out, M);
}
